// MixedFFN_34076270526847
// MI455X (gfx1250) — matrix operand order repaired and checked by reading, not compiled, not run on hardware
//
#include <hip/hip_runtime.h>
#include <hip/hip_bf16.h>
#include <math.h>

typedef __attribute__((ext_vector_type(16))) __bf16 v16bf;
typedef __attribute__((ext_vector_type(8)))  float  v8f;

constexpr int Tt  = 2048;   // tokens per batch
constexpr int Dd  = 1024;   // model dim
constexpr int Ff  = 4096;   // ffn dim
constexpr int LNSc = 64;    // non-shared positions
constexpr int SH  = Tt - LNSc;   // 1984 shared tokens per batch
constexpr int MSH = 8 * SH;      // 15872 shared rows total

__device__ __forceinline__ float gelu_exact(float v) {
    return 0.5f * v * (1.0f + erff(v * 0.70710678118654752f));
}

// ---- fragment layouts used here (A/B in contiguous k order), wave32 ----
// A (16x32 bf16): lane = (m&15) | (((k>>4)&1)<<4) ; idx = k & 15
// B (32x16 bf16): lane = (n&15) | (((k>>4)&1)<<4) ; idx = k & 15
// C (16x16 f32):  vgpr r -> M = r + 8*(lane>>4), N = lane & 15
__device__ __forceinline__ int a_lane(int m, int k) { return (m & 15) | (((k >> 4) & 1) << 4); }
__device__ __forceinline__ int a_idx (int k)        { return k & 15; }

// =====================================================================
// Shared-weight GEMM.  MODE 0:  h = gelu(x @ W1S)      (A=f32 x, C=bf16 h)
//                      MODE 1:  y = h @ W2S            (A=bf16 h, C=f32 out)
// WG tile 128(M) x 128(N); 8 waves in a 4x2 grid, each wave 32x64
// (2 M-frags x 4 N-frags = 8 WMMA per K-step, 1.5 ds_load_b128 per WMMA).
// =====================================================================
template<int MODE>
__global__ __launch_bounds__(256) void ffn_gemm_big(const void* __restrict__ Aptr,
                                                    const float* __restrict__ Bw,
                                                    void* __restrict__ Cptr)
{
    constexpr int Ktot = (MODE == 0) ? Dd : Ff;
    constexpr int Ntot = (MODE == 0) ? Ff : Dd;

    __shared__ __attribute__((aligned(32))) __bf16 As[2][8][32][16];   // 16 KB
    __shared__ __attribute__((aligned(32))) __bf16 Bs[2][8][32][16];   // 16 KB

    const int tid  = threadIdx.x;
    const int lane = tid & 31;
    const int wave = tid >> 5;
    const int wm   = wave >> 1;    // 0..3  -> 32 rows each
    const int wn   = wave & 1;     // 0..1  -> 64 cols each

    const int m0 = blockIdx.x * 128;
    const int n0 = blockIdx.y * 128;

    v8f acc[2][4] = {};

    // A tile 128x32: thread stages one row-half (16 consecutive k).
    auto stageA = [&](int kk, int buf) {
        const int row   = tid >> 1;          // 0..127
        const int kbase = (tid & 1) * 16;
        const int m  = m0 + row;
        const int mb = row >> 4;
        if constexpr (MODE == 0) {
            const int bb = m / SH, ttok = m % SH;
            const float* src = &((const float*)Aptr)[((size_t)bb * Tt + ttok) * (size_t)Dd + kk + kbase];
            #pragma unroll
            for (int j = 0; j < 16; ++j) {
                const int k = kbase + j;
                As[buf][mb][a_lane(row, k)][a_idx(k)] = (__bf16)src[j];
            }
        } else {
            const __bf16* src = &((const __bf16*)Aptr)[(size_t)m * Ff + kk + kbase];
            #pragma unroll
            for (int j = 0; j < 16; ++j) {
                const int k = kbase + j;
                As[buf][mb][a_lane(row, k)][a_idx(k)] = src[j];
            }
        }
    };

    // B tile 32x128: thread stages one half-column (16 consecutive k, one n).
    // Wave-coalesced global reads (32 consecutive n per instruction) and a
    // single contiguous 32-byte LDS store per thread.
    auto stageB = [&](int kk, int buf) {
        const int nl = tid & 127;            // column within tile
        const int kh = tid >> 7;             // k half: 0 -> k 0..15, 1 -> k 16..31
        const float* src = &Bw[(size_t)(kk + kh * 16) * Ntot + n0 + nl];
        v16bf tmp;
        #pragma unroll
        for (int i = 0; i < 16; ++i)
            tmp[i] = (__bf16)src[(size_t)i * Ntot];
        *(v16bf*)(&Bs[buf][nl >> 4][(nl & 15) | (kh << 4)][0]) = tmp;
    };

    stageA(0, 0); stageB(0, 0);
    __syncthreads();
    for (int kk = 0; kk < Ktot; kk += 32) {
        const int buf = (kk >> 5) & 1;
        if (kk + 32 < Ktot) { stageA(kk + 32, buf ^ 1); stageB(kk + 32, buf ^ 1); }

        v16bf af[2], bfr[4];
        #pragma unroll
        for (int mi = 0; mi < 2; ++mi)
            af[mi] = *(const v16bf*)(&As[buf][wm * 2 + mi][lane][0]);
        #pragma unroll
        for (int nt = 0; nt < 4; ++nt)
            bfr[nt] = *(const v16bf*)(&Bs[buf][wn * 4 + nt][lane][0]);
        #pragma unroll
        for (int mi = 0; mi < 2; ++mi)
            #pragma unroll
            for (int nt = 0; nt < 4; ++nt)
                acc[mi][nt] = __builtin_amdgcn_wmma_f32_16x16x32_bf16(
                    false, af[mi], false, bfr[nt], (short)0, acc[mi][nt], false, false);
        __syncthreads();
    }

    #pragma unroll
    for (int mi = 0; mi < 2; ++mi) {
        #pragma unroll
        for (int nt = 0; nt < 4; ++nt) {
            #pragma unroll
            for (int r = 0; r < 8; ++r) {
                const int ml = wm * 32 + mi * 16 + r + ((lane >> 4) << 3);
                const int nl = wn * 64 + nt * 16 + (lane & 15);
                const int m = m0 + ml, n = n0 + nl;
                const float v = acc[mi][nt][r];
                if constexpr (MODE == 0) {
                    ((__bf16*)Cptr)[(size_t)m * Ff + n] = (__bf16)gelu_exact(v);
                } else {
                    const int bb = m / SH, ttok = m % SH;
                    ((float*)Cptr)[((size_t)bb * Tt + ttok) * Dd + n] = v;
                }
            }
        }
    }
}

// =====================================================================
// Non-shared per-position GEMM (M = 8 batches padded to a 16-row tile).
// MODE 0: hns[t] = gelu(x[:,SH+t,:] @ W1NS[t])   (f32 -> bf16 ws)
// MODE 1: yT[t]  = hns[t] @ W2NS[t]              (bf16 -> f32 out)
// Memory-bound: streams 2.15 GB of per-position weights.
// =====================================================================
template<int MODE>
__global__ __launch_bounds__(256) void ffn_gemm_ns(const void* __restrict__ Aptr,
                                                   const float* __restrict__ Wbase,
                                                   void* __restrict__ Cptr)
{
    constexpr int Ktot = (MODE == 0) ? Dd : Ff;
    constexpr int Ntot = (MODE == 0) ? Ff : Dd;

    __shared__ __attribute__((aligned(32))) __bf16 As[2][32][16];      // 2 KB
    __shared__ __attribute__((aligned(32))) __bf16 Bs[2][8][32][16];   // 16 KB

    const int tid  = threadIdx.x;
    const int lane = tid & 31;
    const int wave = tid >> 5;

    const int tpos = blockIdx.x;          // 0..63
    const int n0   = blockIdx.y * 128;
    const float* Bw = Wbase + (size_t)tpos * (size_t)Ktot * Ntot;

    v8f acc = {};

    auto stageA = [&](int kk, int buf) {
        const int row = tid >> 4;         // 0..15 (8..15 are zero pad)
        const int k0  = (tid & 15) * 2;
        #pragma unroll
        for (int j = 0; j < 2; ++j) {
            const int k = k0 + j;
            float v = 0.0f;
            if (row < 8) {
                if constexpr (MODE == 0) {
                    v = ((const float*)Aptr)[((size_t)row * Tt + (SH + tpos)) * (size_t)Dd + kk + k];
                } else {
                    v = (float)((const __bf16*)Aptr)[((size_t)tpos * 8 + row) * (size_t)Ff + kk + k];
                }
            }
            As[buf][a_lane(row, k)][a_idx(k)] = (__bf16)v;
        }
    };

    auto stageB = [&](int kk, int buf) {
        const int nl = tid & 127;
        const int kh = tid >> 7;
        const float* src = &Bw[(size_t)(kk + kh * 16) * Ntot + n0 + nl];
        v16bf tmp;
        #pragma unroll
        for (int i = 0; i < 16; ++i)
            tmp[i] = (__bf16)src[(size_t)i * Ntot];
        *(v16bf*)(&Bs[buf][nl >> 4][(nl & 15) | (kh << 4)][0]) = tmp;
    };

    stageA(0, 0); stageB(0, 0);
    __syncthreads();
    for (int kk = 0; kk < Ktot; kk += 32) {
        const int buf = (kk >> 5) & 1;
        if (kk + 32 < Ktot) { stageA(kk + 32, buf ^ 1); stageB(kk + 32, buf ^ 1); }
        v16bf af = *(const v16bf*)(&As[buf][lane][0]);
        v16bf bf = *(const v16bf*)(&Bs[buf][wave][lane][0]);
        acc = __builtin_amdgcn_wmma_f32_16x16x32_bf16(
            false, af, false, bf, (short)0, acc, false, false);
        __syncthreads();
    }

    if (lane < 16) {   // only M = 0..7 (the 8 real batch rows) are valid
        #pragma unroll
        for (int r = 0; r < 8; ++r) {
            const int n = n0 + wave * 16 + (lane & 15);
            const float v = acc[r];
            if constexpr (MODE == 0) {
                ((__bf16*)Cptr)[((size_t)tpos * 8 + r) * (size_t)Ff + n] = (__bf16)gelu_exact(v);
            } else {
                ((float*)Cptr)[((size_t)r * Tt + SH + tpos) * (size_t)Dd + n] = v;
            }
        }
    }
}

extern "C" void kernel_launch(void* const* d_in, const int* in_sizes, int n_in,
                              void* d_out, int out_size, void* d_ws, size_t ws_size,
                              hipStream_t stream)
{
    const float* x    = (const float*)d_in[0];
    const float* W1S  = (const float*)d_in[1];
    const float* W2S  = (const float*)d_in[2];
    const float* W1NS = (const float*)d_in[3];
    const float* W2NS = (const float*)d_in[4];

    // workspace: h (15872 x 4096 bf16 = 130 MB), then hns (512 x 4096 bf16 = 4 MB)
    __bf16* h   = (__bf16*)d_ws;
    __bf16* hns = (__bf16*)((char*)d_ws + (size_t)MSH * Ff * sizeof(__bf16));

    // shared path: h = gelu(x @ W1S) ; yS = h @ W2S
    ffn_gemm_big<0><<<dim3(MSH / 128, Ff / 128), 256, 0, stream>>>(x, W1S, h);
    // non-shared path (independent of pass 2, streams the big weight tensors)
    ffn_gemm_ns<0><<<dim3(LNSc, Ff / 128), 256, 0, stream>>>(x, W1NS, hns);
    ffn_gemm_big<1><<<dim3(MSH / 128, Dd / 128), 256, 0, stream>>>(h, W2S, (void*)d_out);
    ffn_gemm_ns<1><<<dim3(LNSc, Dd / 128), 256, 0, stream>>>(hns, W2NS, (void*)d_out);
}
